// GroupedQueryAttention_21131239097212
// MI455X (gfx1250) — matrix operand order repaired and checked by reading, not compiled, not run on hardware
//
#include <hip/hip_runtime.h>

// ---------------------------------------------------------------------------
// GQA forward for MI455X (gfx1250, wave32, WMMA 16x16x32 bf16, f32 accum)
// Async global->LDS staging for non-transposing tile copies.
// ---------------------------------------------------------------------------

typedef __attribute__((ext_vector_type(16))) __bf16 bf16x16;
typedef __attribute__((ext_vector_type(8)))  __bf16 bf16x8;
typedef __attribute__((ext_vector_type(8)))  float  floatx8;
typedef unsigned short u16;
typedef int v4i __attribute__((vector_size(16)));

#define DM   2048            // d_model
#define NH   32              // query heads
#define NKV  8               // kv heads
#define DK   64              // head dim
#define BB   2               // batch
#define SS   2048            // seq len
#define MROWS (BB*SS)        // 4096 token rows
#define KVW  (NKV*DK)        // 512

// ---- gfx1250 async global->LDS path (ASYNCcnt) --------------------------
#if defined(__HIP_DEVICE_COMPILE__) &&                                      \
    __has_builtin(__builtin_amdgcn_global_load_async_to_lds_b128) &&        \
    __has_builtin(__builtin_amdgcn_s_wait_asynccnt)
#define HAVE_ASYNC_LDS 1
#else
#define HAVE_ASYNC_LDS 0
#endif

#if HAVE_ASYNC_LDS
__device__ __forceinline__ void async_b128(u16* lds_dst, const u16* gsrc) {
  __builtin_amdgcn_global_load_async_to_lds_b128(
      (__attribute__((address_space(1))) v4i*)gsrc,
      (__attribute__((address_space(3))) v4i*)lds_dst,
      0, 0);
}
#define ASYNC_WAIT() __builtin_amdgcn_s_wait_asynccnt(0)
#else
#define ASYNC_WAIT() ((void)0)
#endif

__device__ __forceinline__ u16 f2bf(float f) {
  unsigned int u = __float_as_uint(f);
  u += 0x7fffu + ((u >> 16) & 1u);      // round-to-nearest-even
  return (u16)(u >> 16);
}

// A-matrix 16x32 bf16 fragment (per ISA 7.12.2):
//   lane L (row = L&15, half = L>>4): j<8 -> k = half*8 + j ; j>=8 -> k = 16 + half*8 + (j-8)
// caller passes pointer at (row*stride + kbase + half*8)
__device__ __forceinline__ bf16x16 frag_a8(const u16* p) {
  bf16x8 lo = *(const bf16x8*)(const void*)p;
  bf16x8 hi = *(const bf16x8*)(const void*)(p + 16);
  return __builtin_shufflevector(lo, hi, 0,1,2,3,4,5,6,7,8,9,10,11,12,13,14,15);
}

// B-matrix 32x16 bf16 fragment: lane holds col N = lane&15, j<8 -> k = half*8 + j ; j>=8 -> k = 16 + half*8 + (j-8)
// caller passes pointer at (col*stride + kbase + half*8); data must be [col][k] in LDS
__device__ __forceinline__ bf16x16 frag_b16(const u16* p) {
  bf16x8 lo = *(const bf16x8*)(const void*)p;
  bf16x8 hi = *(const bf16x8*)(const void*)(p + 16);
  return __builtin_shufflevector(lo, hi, 0,1,2,3,4,5,6,7,8,9,10,11,12,13,14,15);
}

__device__ __forceinline__ floatx8 wmma_bf16(bf16x16 a, bf16x16 b, floatx8 c) {
  return __builtin_amdgcn_wmma_f32_16x16x32_bf16(false, a, false, b, (short)0, c, false, false);
}

// ---------------------------------------------------------------------------
// fp32 -> bf16 conversion
// ---------------------------------------------------------------------------
__global__ void f32_to_bf16_kernel(const float* __restrict__ in,
                                   u16* __restrict__ out, int n) {
  for (int i = blockIdx.x * blockDim.x + threadIdx.x; i < n;
       i += gridDim.x * blockDim.x)
    out[i] = f2bf(in[i]);
}

// ---------------------------------------------------------------------------
// Tiled bf16 WMMA GEMM: C[M,N] = A[M,K] * B[K,N] + bias[N]
// block = 128x128, 256 threads (8 waves, 2x4), wave tile = 64x32 (4x2 WMMA)
// ---------------------------------------------------------------------------
template <bool OUT_F32>
__global__ __launch_bounds__(256) void gemm_bf16_wmma(
    const u16* __restrict__ A, const u16* __restrict__ B,
    const float* __restrict__ bias, void* __restrict__ Cout,
    int M, int N, int Kd) {
  constexpr int AS = 40;   // LDS row stride (bf16 elems) for A tile [128][32]
  constexpr int BS = 40;   // LDS col stride for B tile stored transposed [128 cols][32 k]
  __shared__ u16 ldsA[128 * AS];
  __shared__ u16 ldsB[128 * BS];

  const int t    = threadIdx.x;
  const int wave = t >> 5, lane = t & 31;
  const int half = lane >> 4, l15 = lane & 15;
  const int wm = wave >> 2, wn = wave & 3;          // 2 x 4 wave grid
  const int mb = blockIdx.y, nb = blockIdx.x;

  floatx8 zf = {};
  floatx8 acc[4][2];
#pragma unroll
  for (int mt = 0; mt < 4; ++mt)
#pragma unroll
    for (int nt = 0; nt < 2; ++nt) acc[mt][nt] = zf;

  const int arow = t >> 1, akoff = (t & 1) * 16;    // A: 128 rows x 32 k
  const int brow = t >> 3, bcoff = (t & 7) * 16;    // B: 32 k-rows x 128 cols

  for (int k0 = 0; k0 < Kd; k0 += 32) {
    __syncthreads();
    // ---- stage A tile (row-major, straight copy -> async to LDS) ----
    {
      const u16* gsrc = A + (size_t)(mb * 128 + arow) * Kd + k0 + akoff;
      u16* ldst = ldsA + arow * AS + akoff;
#if HAVE_ASYNC_LDS
      async_b128(ldst, gsrc);
      async_b128(ldst + 8, gsrc + 8);
#else
      const uint4* src = (const uint4*)gsrc;
      uint4* dst = (uint4*)ldst;
      dst[0] = src[0];
      dst[1] = src[1];
#endif
    }
    // ---- stage B tile transposed: ldsB[col][k] (VGPR unpack) ----
    {
      const uint4* src =
          (const uint4*)(B + (size_t)(k0 + brow) * N + nb * 128 + bcoff);
      union { uint4 v; u16 s[8]; } u0, u1;
      u0.v = src[0];
      u1.v = src[1];
#pragma unroll
      for (int i = 0; i < 8; ++i) {
        ldsB[(bcoff + i) * BS + brow]     = u0.s[i];
        ldsB[(bcoff + 8 + i) * BS + brow] = u1.s[i];
      }
    }
    if (k0 + 32 < Kd) {
      __builtin_prefetch(A + (size_t)(mb * 128 + arow) * Kd + k0 + 32 + akoff, 0, 1);
      __builtin_prefetch(B + (size_t)(k0 + 32 + brow) * N + nb * 128 + bcoff, 0, 1);
    }
    ASYNC_WAIT();
    __syncthreads();

    bf16x16 af[4], bfv[2];
#pragma unroll
    for (int mt = 0; mt < 4; ++mt)
      af[mt] = frag_a8(ldsA + (wm * 64 + mt * 16 + l15) * AS + half * 8);
#pragma unroll
    for (int nt = 0; nt < 2; ++nt)
      bfv[nt] = frag_b16(ldsB + (wn * 32 + nt * 16 + l15) * BS + half * 8);
#pragma unroll
    for (int mt = 0; mt < 4; ++mt)
#pragma unroll
      for (int nt = 0; nt < 2; ++nt)
        acc[mt][nt] = wmma_bf16(af[mt], bfv[nt], acc[mt][nt]);
  }

  // ---- store: C tile lane layout: row = v + 8*half, col = l15 ----
#pragma unroll
  for (int mt = 0; mt < 4; ++mt)
#pragma unroll
    for (int nt = 0; nt < 2; ++nt)
#pragma unroll
      for (int v = 0; v < 8; ++v) {
        const int row = mb * 128 + wm * 64 + mt * 16 + v + 8 * half;
        const int col = nb * 128 + wn * 32 + nt * 16 + l15;
        float val = acc[mt][nt][v] + (bias ? bias[col] : 0.0f);
        if (OUT_F32)
          ((float*)Cout)[(size_t)row * N + col] = val;
        else
          ((u16*)Cout)[(size_t)row * N + col] = f2bf(val);
      }
}

// ---------------------------------------------------------------------------
// Flash-attention GQA kernel.
// grid = (S/128, B*NH); block = 256 (8 waves). Each wave owns a 16-row strip
// of the 128-query block. Key blocks of 128, causal.
// ---------------------------------------------------------------------------
constexpr int QS_STRIDE = 72;    // [128][64] Q tile,  +8 pad
constexpr int KS_STRIDE = 72;    // [128][64] K tile,  +8 pad
constexpr int VS_STRIDE = 136;   // [64][128] V tile transposed [d][key]
constexpr int PS_STRIDE = 136;   // [128][128] P (probs) tile
constexpr size_t ATTN_SMEM_BYTES =
    (size_t)(128 * QS_STRIDE + 128 * KS_STRIDE + 64 * VS_STRIDE +
             128 * PS_STRIDE) * 2;  // 89088 bytes

__global__ __launch_bounds__(256) void gqa_attn_kernel(
    const u16* __restrict__ Q, const u16* __restrict__ K,
    const u16* __restrict__ V, u16* __restrict__ O) {
  extern __shared__ u16 smem[];
  u16* sQ = smem;                       // 128 x QS_STRIDE  [row][d]
  u16* sK = sQ + 128 * QS_STRIDE;       // 128 x KS_STRIDE  [key][d]
  u16* sV = sK + 128 * KS_STRIDE;       // 64  x VS_STRIDE  [d][key]
  u16* sP = sV + 64 * VS_STRIDE;        // 128 x PS_STRIDE  [row][key]

  const int t    = threadIdx.x;
  const int wave = t >> 5, lane = t & 31;
  const int half = lane >> 4, l15 = lane & 15;
  const int qb = blockIdx.x, bh = blockIdx.y;
  const int bi = bh >> 5, h = bh & 31, kvh = h >> 2;  // g = NH/NKV = 4
  const int q0 = qb * 128;

  const u16* Qbase = Q + (size_t)bi * SS * DM + h * DK;     // row stride DM
  const u16* Kbase = K + (size_t)bi * SS * KVW + kvh * DK;  // row stride KVW
  const u16* Vbase = V + (size_t)bi * SS * KVW + kvh * DK;

  // ---- stage Q block: 128 rows x 64 d (straight copy -> async) ----
  {
    const int row = t >> 1, c0 = (t & 1) * 32;
    const u16* gsrc = Qbase + (size_t)(q0 + row) * DM + c0;
    u16* ldst = sQ + row * QS_STRIDE + c0;
#if HAVE_ASYNC_LDS
#pragma unroll
    for (int j = 0; j < 4; ++j) async_b128(ldst + j * 8, gsrc + j * 8);
#else
    const uint4* src = (const uint4*)gsrc;
    uint4* dst = (uint4*)ldst;
    dst[0] = src[0]; dst[1] = src[1]; dst[2] = src[2]; dst[3] = src[3];
#endif
  }
  ASYNC_WAIT();
  __syncthreads();

  // Q fragments for this wave's 16-row strip (d split into 2 k-steps of 32)
  bf16x16 aq[2];
#pragma unroll
  for (int dh = 0; dh < 2; ++dh)
    aq[dh] = frag_a8(sQ + (wave * 16 + l15) * QS_STRIDE + dh * 32 + half * 8);

  floatx8 zf = {};
  floatx8 ov[4];            // O accum: 16 rows x 64 d = 4 n-tiles
#pragma unroll
  for (int dt = 0; dt < 4; ++dt) ov[dt] = zf;
  float mrun[8], lrun[8];
#pragma unroll
  for (int v = 0; v < 8; ++v) { mrun[v] = -3.0e38f; lrun[v] = 0.0f; }

  for (int kb = 0; kb <= qb; ++kb) {
    __syncthreads();  // previous iteration finished reading sK/sV/sP
    // ---- stage K tile [key][d] (straight copy -> async) ----
    {
      const int row = t >> 1, c0 = (t & 1) * 32;
      const u16* gsrc = Kbase + (size_t)(kb * 128 + row) * KVW + c0;
      u16* ldst = sK + row * KS_STRIDE + c0;
#if HAVE_ASYNC_LDS
#pragma unroll
      for (int j = 0; j < 4; ++j) async_b128(ldst + j * 8, gsrc + j * 8);
#else
      const uint4* src = (const uint4*)gsrc;
      uint4* dst = (uint4*)ldst;
      dst[0] = src[0]; dst[1] = src[1]; dst[2] = src[2]; dst[3] = src[3];
#endif
    }
    // ---- stage V tile transposed [d][key] (VGPR unpack) ----
    {
      const int key = t >> 1, d0 = (t & 1) * 32;
      const uint4* src =
          (const uint4*)(Vbase + (size_t)(kb * 128 + key) * KVW + d0);
#pragma unroll
      for (int j = 0; j < 4; ++j) {
        union { uint4 v; u16 s[8]; } u;
        u.v = src[j];
#pragma unroll
        for (int i = 0; i < 8; ++i) sV[(d0 + j * 8 + i) * VS_STRIDE + key] = u.s[i];
      }
    }
    ASYNC_WAIT();
    __syncthreads();

    // ---- scores: S = Q Kt, 16 rows x 128 keys per wave ----
    floatx8 sc[8];
#pragma unroll
    for (int nt = 0; nt < 8; ++nt) sc[nt] = zf;
#pragma unroll
    for (int dh = 0; dh < 2; ++dh)
#pragma unroll
      for (int nt = 0; nt < 8; ++nt) {
        // B = Kt slice: b[j] = K[key][d], key=col, k-index=d -> row-major K tile
        bf16x16 bk =
            frag_b16(sK + (nt * 16 + l15) * KS_STRIDE + dh * 32 + half * 8);
        sc[nt] = wmma_bf16(aq[dh], bk, sc[nt]);
      }

    // ---- scale + causal mask + online softmax ----
    const bool diag = (kb == qb);
    float rmax[8];
#pragma unroll
    for (int v = 0; v < 8; ++v) rmax[v] = -3.0e38f;
#pragma unroll
    for (int nt = 0; nt < 8; ++nt)
#pragma unroll
      for (int v = 0; v < 8; ++v) {
        float s = sc[nt][v] * 0.125f;  // 1/sqrt(64)
        if (diag) {
          const int qg = q0 + wave * 16 + v + 8 * half;
          const int kg = kb * 128 + nt * 16 + l15;
          if (kg > qg) s = -3.0e38f;
        }
        sc[nt][v] = s;
        rmax[v] = fmaxf(rmax[v], s);
      }
#pragma unroll
    for (int m = 1; m < 16; m <<= 1)
#pragma unroll
      for (int v = 0; v < 8; ++v)
        rmax[v] = fmaxf(rmax[v], __shfl_xor(rmax[v], m, 32));

    float sclv[8], rsum[8];
#pragma unroll
    for (int v = 0; v < 8; ++v) {
      const float mn = fmaxf(mrun[v], rmax[v]);
      sclv[v] = __expf(mrun[v] - mn);
      mrun[v] = mn;
      rsum[v] = 0.0f;
    }
#pragma unroll
    for (int nt = 0; nt < 8; ++nt)
#pragma unroll
      for (int v = 0; v < 8; ++v) {
        const float pv = __expf(sc[nt][v] - mrun[v]);
        sc[nt][v] = pv;
        rsum[v] += pv;
      }
#pragma unroll
    for (int m = 1; m < 16; m <<= 1)
#pragma unroll
      for (int v = 0; v < 8; ++v) rsum[v] += __shfl_xor(rsum[v], m, 32);
#pragma unroll
    for (int v = 0; v < 8; ++v) lrun[v] = lrun[v] * sclv[v] + rsum[v];
#pragma unroll
    for (int dt = 0; dt < 4; ++dt)
#pragma unroll
      for (int v = 0; v < 8; ++v) ov[dt][v] *= sclv[v];

    // ---- write P (bf16) to LDS strip ----
#pragma unroll
    for (int nt = 0; nt < 8; ++nt)
#pragma unroll
      for (int v = 0; v < 8; ++v)
        sP[(wave * 16 + v + 8 * half) * PS_STRIDE + nt * 16 + l15] =
            f2bf(sc[nt][v]);
    __syncthreads();

    // ---- O += P V : key dim 128 = 4 wmma k-steps ----
#pragma unroll
    for (int kk = 0; kk < 128; kk += 32) {
      bf16x16 ap =
          frag_a8(sP + (wave * 16 + l15) * PS_STRIDE + kk + half * 8);
#pragma unroll
      for (int dt = 0; dt < 4; ++dt) {
        bf16x16 bv2 =
            frag_b16(sV + (dt * 16 + l15) * VS_STRIDE + kk + half * 8);
        ov[dt] = wmma_bf16(ap, bv2, ov[dt]);
      }
    }
  }

  // ---- epilogue: normalize and store attn_out[b, q, h*64 + d] bf16 ----
#pragma unroll
  for (int dt = 0; dt < 4; ++dt)
#pragma unroll
    for (int v = 0; v < 8; ++v) {
      const int row = q0 + wave * 16 + v + 8 * half;
      const float o = ov[dt][v] / lrun[v];
      O[((size_t)bi * SS + row) * DM + h * DK + dt * 16 + l15] = f2bf(o);
    }
}

// ---------------------------------------------------------------------------
// Host-side launcher
// ---------------------------------------------------------------------------
extern "C" void kernel_launch(void* const* d_in, const int* in_sizes, int n_in,
                              void* d_out, int out_size, void* d_ws,
                              size_t ws_size, hipStream_t stream) {
  (void)in_sizes; (void)n_in; (void)out_size; (void)ws_size;
  const float* x  = (const float*)d_in[0];
  // d_in[1] = mask (causal tril) — applied analytically in the kernel
  const float* Wq = (const float*)d_in[2];
  const float* bq = (const float*)d_in[3];
  const float* Wk = (const float*)d_in[4];
  const float* bk = (const float*)d_in[5];
  const float* Wv = (const float*)d_in[6];
  const float* bv = (const float*)d_in[7];
  const float* Wo = (const float*)d_in[8];
  const float* bo = (const float*)d_in[9];
  float* out = (float*)d_out;

  // workspace carve-up (bf16 buffers)
  char* p = (char*)d_ws;
  u16* xb  = (u16*)p; p += (size_t)MROWS * DM * 2;   // x bf16; reused as attn_out
  u16* Wqb = (u16*)p; p += (size_t)DM * DM * 2;
  u16* Wkb = (u16*)p; p += (size_t)DM * KVW * 2;
  u16* Wvb = (u16*)p; p += (size_t)DM * KVW * 2;
  u16* Wob = (u16*)p; p += (size_t)DM * DM * 2;
  u16* Qb  = (u16*)p; p += (size_t)MROWS * DM * 2;
  u16* Kb  = (u16*)p; p += (size_t)MROWS * KVW * 2;
  u16* Vb  = (u16*)p; p += (size_t)MROWS * KVW * 2;
  u16* AOb = xb;  // x is dead after the QKV GEMMs; reuse for attention output

  auto cvt = [&](const float* src, u16* dst, size_t n) {
    int blocks = (int)((n + 256 * 16 - 1) / (256 * 16));
    f32_to_bf16_kernel<<<blocks, 256, 0, stream>>>(src, dst, (int)n);
  };
  cvt(x,  xb,  (size_t)MROWS * DM);
  cvt(Wq, Wqb, (size_t)DM * DM);
  cvt(Wk, Wkb, (size_t)DM * KVW);
  cvt(Wv, Wvb, (size_t)DM * KVW);
  cvt(Wo, Wob, (size_t)DM * DM);

  // QKV projections (bf16 out)
  gemm_bf16_wmma<false><<<dim3(DM / 128, MROWS / 128), 256, 0, stream>>>(
      xb, Wqb, bq, Qb, MROWS, DM, DM);
  gemm_bf16_wmma<false><<<dim3(KVW / 128, MROWS / 128), 256, 0, stream>>>(
      xb, Wkb, bk, Kb, MROWS, KVW, DM);
  gemm_bf16_wmma<false><<<dim3(KVW / 128, MROWS / 128), 256, 0, stream>>>(
      xb, Wvb, bv, Vb, MROWS, KVW, DM);

  // fused causal GQA attention
  (void)hipFuncSetAttribute((const void*)gqa_attn_kernel,
                            hipFuncAttributeMaxDynamicSharedMemorySize,
                            (int)ATTN_SMEM_BYTES);
  gqa_attn_kernel<<<dim3(SS / 128, BB * NH), 256, ATTN_SMEM_BYTES, stream>>>(
      Qb, Kb, Vb, AOb);

  // output projection (f32 out + bias)
  gemm_bf16_wmma<true><<<dim3(DM / 128, MROWS / 128), 256, 0, stream>>>(
      AOb, Wob, bo, out, MROWS, DM, DM);
}
